// DOASNN_48627619725475
// MI455X (gfx1250) — matrix operand order repaired and checked by reading, not compiled, not run on hardware
//
#include <hip/hip_runtime.h>
#include <math.h>

// ---------------------------------------------------------------------------
// Types for CDNA5 WMMA (wave32, 16x16x32 f16 -> f32)
// ---------------------------------------------------------------------------
typedef __attribute__((ext_vector_type(16))) _Float16 v16h;
typedef __attribute__((ext_vector_type(8)))  _Float16 v8h;
typedef __attribute__((ext_vector_type(8)))  float    v8f;

union V16 { v16h v; v8h h[2]; };

// Problem constants
#define BATCH 2048
#define CIN   10
#define TSTEP 33
#define FDIM  48
#define C1    32
#define C2    64
#define HREC  192
#define KP2   768          // 64*12
#define KTOT  960          // 768 + 192
#define DOUT  36
#define NPAD  48           // readout N padded to 3 WMMA tiles

// ---------------------------------------------------------------------------
// Workspace layout (bytes). P2ALL holds pooled conv spikes for all T (f16).
// ---------------------------------------------------------------------------
static constexpr size_t OFF_P2ALL = 0;                                         // 33*2048*768*2
static constexpr size_t OFF_MEMR  = OFF_P2ALL + (size_t)TSTEP*BATCH*KP2*2;     // 2048*192 f32
static constexpr size_t OFF_SPK0  = OFF_MEMR  + (size_t)BATCH*HREC*4;          // 2048*192 f16
static constexpr size_t OFF_SPK1  = OFF_SPK0  + (size_t)BATCH*HREC*2;
static constexpr size_t OFF_MR16  = OFF_SPK1  + (size_t)BATCH*HREC*2;          // 2048*192 f16
static constexpr size_t OFF_LOGI  = OFF_MR16  + (size_t)BATCH*HREC*2;          // 2048*48 f32
static constexpr size_t OFF_SCAL  = OFF_LOGI  + (size_t)BATCH*NPAD*4;          // betas + spike acc
static constexpr size_t OFF_WCAT  = OFF_SCAL  + 256;                           // 192*960 f16
static constexpr size_t OFF_WR    = OFF_WCAT  + (size_t)HREC*KTOT*2;           // 48*192 f16

// ---------------------------------------------------------------------------
// Fragment loaders per CDNA5 ISA VGPR layouts (CDNA5 ISA §7.12.2)
// A 16x32 (MxK): lane l (m = l&15): elems 0..7 = K k0+(l<16?0:8)..+7,
//                elems 8..15 = K k0+16+(l<16?0:8)..+7          (2x b128)
// B 32x16 (KxN): lane l (n = l&15): elems 0..7 = K k0+(l<16?0:8)..+7,
//                elems 8..15 = K k0+16+(l<16?0:8)..+7          (2x b128)
// ---------------------------------------------------------------------------
__device__ inline v16h load_a_frag(const _Float16* A, int lda, int m0, int k0, int lane) {
    int m = m0 + (lane & 15);
    int koff = (lane < 16) ? 0 : 8;
    const _Float16* p = A + (size_t)m * lda + k0 + koff;
    V16 r;
    r.h[0] = *(const v8h*)(p);
    r.h[1] = *(const v8h*)(p + 16);
    return r.v;
}

__device__ inline v16h load_b_frag(const _Float16* W, int ldw, int n0, int k0, int lane) {
    int n = n0 + (lane & 15);
    int koff = (lane < 16) ? 0 : 8;
    const _Float16* p = W + (size_t)n * ldw + k0 + koff;
    V16 r;
    r.h[0] = *(const v8h*)(p);
    r.h[1] = *(const v8h*)(p + 16);
    return r.v;
}

// ---------------------------------------------------------------------------
// Zero a byte span (16B granules)
// ---------------------------------------------------------------------------
__global__ void zero_ws_kernel(uint4* p, size_t n16) {
    size_t i = (size_t)blockIdx.x * blockDim.x + threadIdx.x;
    if (i < n16) p[i] = make_uint4(0u, 0u, 0u, 0u);
}

// ---------------------------------------------------------------------------
// Convert FC weights to f16 (concat [in_fc | rec_fc] along K; padded readout),
// compute clamped sigmoid betas.
// ---------------------------------------------------------------------------
__global__ void init_weights_kernel(const float* __restrict__ in_fc,
                                    const float* __restrict__ rec_fc,
                                    const float* __restrict__ readout_w,
                                    const float* __restrict__ b1l,
                                    const float* __restrict__ b2l,
                                    const float* __restrict__ brl,
                                    _Float16* __restrict__ Wcat,
                                    _Float16* __restrict__ Wr,
                                    float* __restrict__ scalars) {
    int i = blockIdx.x * blockDim.x + threadIdx.x;
    if (i < HREC * KTOT) {
        int n = i / KTOT, k = i % KTOT;
        float v = (k < KP2) ? in_fc[(size_t)n * KP2 + k]
                            : rec_fc[(size_t)n * HREC + (k - KP2)];
        Wcat[i] = (_Float16)v;
    }
    int j = i - HREC * KTOT;
    if (j >= 0 && j < NPAD * HREC) {
        int n = j / HREC, k = j % HREC;
        float v = (n < DOUT) ? readout_w[(size_t)n * HREC + k] : 0.0f;
        Wr[j] = (_Float16)v;
    }
    if (i == 0) {
        float s1 = 1.0f / (1.0f + expf(-b1l[0]));
        float s2 = 1.0f / (1.0f + expf(-b2l[0]));
        float sr = 1.0f / (1.0f + expf(-brl[0]));
        scalars[0] = fminf(fmaxf(s1, 0.5f), 0.999f);
        scalars[1] = fminf(fmaxf(s2, 0.5f), 0.999f);
        scalars[2] = fminf(fmaxf(sr, 0.5f), 0.999f);
    }
}

// ---------------------------------------------------------------------------
// Persistent conv1+LIF1+pool+conv2+LIF2+pool over ALL 33 timesteps.
// One 256-thread workgroup per batch element; membrane state in registers;
// conv weights + activations staged through LDS. Emits p2 (f16) per t.
// ---------------------------------------------------------------------------
__global__ void __launch_bounds__(256)
conv_lif_all_kernel(const float* __restrict__ x,
                    const float* __restrict__ w1g,
                    const float* __restrict__ w2g,
                    _Float16* __restrict__ P2all,
                    const float* __restrict__ scalars,
                    float* __restrict__ spike_accum) {
    __shared__ float w1s[C1 * CIN * 3];     //  3840 B
    __shared__ float w2s[C2 * C1 * 3];      // 24576 B
    __shared__ float xs[CIN][FDIM + 2];     //  2000 B (zero-padded borders)
    __shared__ float p1s[C1][FDIM / 2 + 2]; //  3328 B (zero-padded borders)
    __shared__ float red[256];

    const int tid = threadIdx.x;
    const int b = blockIdx.x;

    // stage weights into LDS once
    for (int i = tid; i < C1 * CIN * 3; i += 256) w1s[i] = w1g[i];
    for (int i = tid; i < C2 * C1 * 3; i += 256) w2s[i] = w2g[i];
    if (tid < CIN) { xs[tid][0] = 0.0f; xs[tid][FDIM + 1] = 0.0f; }
    if (tid < C1)  { p1s[tid][0] = 0.0f; p1s[tid][FDIM / 2 + 1] = 0.0f; }

    const float beta1 = scalars[0];
    const float beta2 = scalars[1];

    // thread ownership (fixed across time)
    const int co1 = tid & 31;          // conv1 out channel
    const int f01 = (tid >> 5) * 6;    // conv1 position block (6 wide)
    const int co2 = tid & 63;          // conv2 out channel
    const int f02 = (tid >> 6) * 6;    // conv2 position block (6 wide)

    float m1[6] = {0, 0, 0, 0, 0, 0};  // membrane state in registers
    float m2[6] = {0, 0, 0, 0, 0, 0};
    float lsp = 0.0f;

    for (int t = 0; t < TSTEP; ++t) {
        __syncthreads();  // p1s free from previous t's stage 2
        // load x_t[b] -> LDS (padded)
        for (int idx = tid; idx < CIN * FDIM; idx += 256) {
            int ci = idx / FDIM, f = idx % FDIM;
            xs[ci][f + 1] = x[(((size_t)b * CIN + ci) * TSTEP + t) * FDIM + f];
        }
        __syncthreads();

        // ---- stage 1: conv1 + LIF1 + pool -> p1s ----
        {
            float cur[6] = {0, 0, 0, 0, 0, 0};
            for (int ci = 0; ci < CIN; ++ci) {
                const float* wp = &w1s[(co1 * CIN + ci) * 3];
                float wa = wp[0], wb = wp[1], wc = wp[2];
                const float* xr = &xs[ci][f01];
#pragma unroll
                for (int j = 0; j < 6; ++j)
                    cur[j] += wa * xr[j] + wb * xr[j + 1] + wc * xr[j + 2];
            }
            float s[6];
#pragma unroll
            for (int j = 0; j < 6; ++j) {
                float m = beta1 * m1[j] + cur[j];
                float sv = (m > 1.0f) ? 1.0f : 0.0f;
                m1[j] = m - sv;
                s[j] = sv;
                lsp += sv;
            }
            int p0 = f01 >> 1;
            p1s[co1][1 + p0]     = 0.5f * (s[0] + s[1]);
            p1s[co1][1 + p0 + 1] = 0.5f * (s[2] + s[3]);
            p1s[co1][1 + p0 + 2] = 0.5f * (s[4] + s[5]);
        }
        __syncthreads();

        // ---- stage 2: conv2 + LIF2 + pool -> P2all[t] ----
        {
            float cur[6] = {0, 0, 0, 0, 0, 0};
            for (int ci = 0; ci < C1; ++ci) {
                const float* wp = &w2s[(co2 * C1 + ci) * 3];
                float wa = wp[0], wb = wp[1], wc = wp[2];
                const float* pr = &p1s[ci][f02];
#pragma unroll
                for (int j = 0; j < 6; ++j)
                    cur[j] += wa * pr[j] + wb * pr[j + 1] + wc * pr[j + 2];
            }
            float s[6];
#pragma unroll
            for (int j = 0; j < 6; ++j) {
                float m = beta2 * m2[j] + cur[j];
                float sv = (m > 1.0f) ? 1.0f : 0.0f;
                m2[j] = m - sv;
                s[j] = sv;
                lsp += sv;
            }
            int p0 = f02 >> 1;  // 0,3,6,9
            size_t abase = ((size_t)t * BATCH + b) * KP2 + (size_t)co2 * 12 + p0;
            P2all[abase]     = (_Float16)(0.5f * (s[0] + s[1]));
            P2all[abase + 1] = (_Float16)(0.5f * (s[2] + s[3]));
            P2all[abase + 2] = (_Float16)(0.5f * (s[4] + s[5]));
        }
    }

    // block-level spike reduction, one atomic per block
    red[tid] = lsp;
    __syncthreads();
    for (int off = 128; off > 0; off >>= 1) {
        if (tid < off) red[tid] += red[tid + off];
        __syncthreads();
    }
    if (tid == 0) atomicAdd(spike_accum, red[0]);
}

// ---------------------------------------------------------------------------
// Recurrent FC via WMMA: cur_r = [p2 | spk_prev] @ [in_fc | rec_fc]^T
// (M=2048, N=192, K=960 = 24 p2-tiles + 6 spk-tiles), fused LIF_r epilogue.
// One wave per 16x16 output tile; 30x v_wmma_f32_16x16x32_f16 per wave.
// ---------------------------------------------------------------------------
__global__ void __launch_bounds__(128)
fc_wmma_kernel(const _Float16* __restrict__ P2t,      // [B, 768]
               const _Float16* __restrict__ spkPrev,  // [B, 192]
               const _Float16* __restrict__ Wcat,     // [192, 960]
               float* __restrict__ memr,              // [B, 192] f32 state
               _Float16* __restrict__ memr16,         // [B, 192] post-update
               _Float16* __restrict__ spkNext,        // [B, 192]
               const float* __restrict__ scalars,
               float* __restrict__ spike_accum) {
    const int lane = threadIdx.x & 31;
    const int wave = blockIdx.x * (blockDim.x >> 5) + (threadIdx.x >> 5);
    const int mtile = wave / 12;
    const int ntile = wave % 12;
    const int m0 = mtile * 16;
    const int n0 = ntile * 16;

    v8f acc = {};
#pragma unroll 2
    for (int kt = 0; kt < 30; ++kt) {
        v16h a = (kt < 24) ? load_a_frag(P2t, KP2, m0, kt * 32, lane)
                           : load_a_frag(spkPrev, HREC, m0, (kt - 24) * 32, lane);
        v16h bm = load_b_frag(Wcat, KTOT, n0, kt * 32, lane);
        acc = __builtin_amdgcn_wmma_f32_16x16x32_f16(
            false, a, false, bm, (short)0, acc, false, false);
    }

    // fused LIF_r epilogue; C layout: elem r -> (M = m0 + r + (lane<16?0:8), N = n0 + lane&15)
    const float beta_r = scalars[2];
    const int n = n0 + (lane & 15);
    const int mbase = m0 + ((lane < 16) ? 0 : 8);
    float lsp = 0.0f;
#pragma unroll
    for (int r = 0; r < 8; ++r) {
        size_t idx = (size_t)(mbase + r) * HREC + n;
        float mem = beta_r * memr[idx] + acc[r];
        float sv = (mem > 1.0f) ? 1.0f : 0.0f;
        mem -= sv;
        memr[idx] = mem;
        memr16[idx] = (_Float16)mem;
        spkNext[idx] = (_Float16)sv;
        lsp += sv;
    }
    for (int off = 16; off > 0; off >>= 1) lsp += __shfl_xor(lsp, off, 32);
    if (lane == 0) atomicAdd(spike_accum, lsp);
}

// ---------------------------------------------------------------------------
// Readout via WMMA: logits += mem_r @ readout_w^T  (M=2048, N=48 pad, K=192)
// ---------------------------------------------------------------------------
__global__ void __launch_bounds__(128)
readout_wmma_kernel(const _Float16* __restrict__ memr16,  // [B, 192]
                    const _Float16* __restrict__ Wr,      // [48, 192]
                    float* __restrict__ logits) {         // [B, 48]
    const int lane = threadIdx.x & 31;
    const int wave = blockIdx.x * (blockDim.x >> 5) + (threadIdx.x >> 5);
    const int mtile = wave / 3;
    const int ntile = wave % 3;
    const int m0 = mtile * 16;
    const int n0 = ntile * 16;

    v8f acc = {};
#pragma unroll
    for (int kt = 0; kt < 6; ++kt) {
        v16h a = load_a_frag(memr16, HREC, m0, kt * 32, lane);
        v16h bm = load_b_frag(Wr, HREC, n0, kt * 32, lane);
        acc = __builtin_amdgcn_wmma_f32_16x16x32_f16(
            false, a, false, bm, (short)0, acc, false, false);
    }
    const int n = n0 + (lane & 15);
    const int mbase = m0 + ((lane < 16) ? 0 : 8);
#pragma unroll
    for (int r = 0; r < 8; ++r) {
        size_t idx = (size_t)(mbase + r) * NPAD + n;
        logits[idx] += acc[r];
    }
}

// ---------------------------------------------------------------------------
// Finalize: out[b,36] = logits/T + bias ; out[last] = spikes / total_neurons
// ---------------------------------------------------------------------------
__global__ void finalize_kernel(const float* __restrict__ logits,
                                const float* __restrict__ rb,
                                const float* __restrict__ scalars,
                                float* __restrict__ out) {
    int i = blockIdx.x * blockDim.x + threadIdx.x;
    if (i < BATCH * DOUT) {
        int b = i / DOUT, j = i % DOUT;
        out[i] = logits[(size_t)b * NPAD + j] * (1.0f / (float)TSTEP) + rb[j];
    } else if (i == BATCH * DOUT) {
        // total_neurons = T * B * (32*48 + 64*24 + 192) = 33*2048*3264
        out[i] = scalars[3] * (1.0f / 220594176.0f);
    }
}

// ---------------------------------------------------------------------------
// Host launcher (graph-capture safe: kernels only, all on `stream`)
// ---------------------------------------------------------------------------
extern "C" void kernel_launch(void* const* d_in, const int* in_sizes, int n_in,
                              void* d_out, int out_size, void* d_ws, size_t ws_size,
                              hipStream_t stream) {
    (void)in_sizes; (void)n_in; (void)out_size; (void)ws_size;
    const float* x         = (const float*)d_in[0];
    const float* conv1_w   = (const float*)d_in[1];
    const float* conv2_w   = (const float*)d_in[2];
    const float* in_fc_w   = (const float*)d_in[3];
    const float* rec_fc_w  = (const float*)d_in[4];
    const float* readout_w = (const float*)d_in[5];
    const float* readout_b = (const float*)d_in[6];
    const float* b1l       = (const float*)d_in[7];
    const float* b2l       = (const float*)d_in[8];
    const float* brl       = (const float*)d_in[9];

    char* ws = (char*)d_ws;
    _Float16* P2all  = (_Float16*)(ws + OFF_P2ALL);
    float*    memr   = (float*)   (ws + OFF_MEMR);
    _Float16* spk0   = (_Float16*)(ws + OFF_SPK0);
    _Float16* spk1   = (_Float16*)(ws + OFF_SPK1);
    _Float16* memr16 = (_Float16*)(ws + OFF_MR16);
    float*    logits = (float*)   (ws + OFF_LOGI);
    float*    scal   = (float*)   (ws + OFF_SCAL);
    _Float16* Wcat   = (_Float16*)(ws + OFF_WCAT);
    _Float16* Wr     = (_Float16*)(ws + OFF_WR);
    float*    spikes = scal + 3;

    // zero persistent state region [OFF_MEMR, OFF_WCAT)
    size_t n16 = (OFF_WCAT - OFF_MEMR) / 16;
    zero_ws_kernel<<<(unsigned)((n16 + 255) / 256), 256, 0, stream>>>(
        (uint4*)(ws + OFF_MEMR), n16);

    int wi_n = HREC * KTOT + NPAD * HREC;
    init_weights_kernel<<<(wi_n + 255) / 256, 256, 0, stream>>>(
        in_fc_w, rec_fc_w, readout_w, b1l, b2l, brl, Wcat, Wr, scal);

    // all-timestep conv/LIF pipeline (state lives in registers)
    conv_lif_all_kernel<<<BATCH, 256, 0, stream>>>(
        x, conv1_w, conv2_w, P2all, scal, spikes);

    // sequential recurrence: FC-WMMA + readout-WMMA per timestep
    for (int t = 0; t < TSTEP; ++t) {
        const _Float16* P2t = P2all + (size_t)t * BATCH * KP2;
        _Float16* sPrev = (t & 1) ? spk1 : spk0;
        _Float16* sNext = (t & 1) ? spk0 : spk1;
        fc_wmma_kernel<<<(BATCH / 16) * 12 / 4, 128, 0, stream>>>(
            P2t, sPrev, Wcat, memr, memr16, sNext, scal, spikes);
        readout_wmma_kernel<<<(BATCH / 16) * 3 / 4, 128, 0, stream>>>(
            memr16, Wr, logits);
    }

    finalize_kernel<<<(BATCH * DOUT + 1 + 255) / 256, 256, 0, stream>>>(
        logits, readout_b, scal, (float*)d_out);
}
